// TreeDecoder_57595511439976
// MI455X (gfx1250) — compile-verified, matrix operand order checked by reading, not run on hardware
//
#include <hip/hip_runtime.h>
#include <hip/hip_bf16.h>

typedef __attribute__((ext_vector_type(16))) __bf16  v16bf;
typedef __attribute__((ext_vector_type(8)))  float   v8f;
typedef __attribute__((ext_vector_type(4)))  unsigned int v4u;
typedef unsigned short u16b;

union Frag {
    v16bf v;
    u16b  u[16];
    v4u   q[2];
};

// K index held by (lane, element) for the 16-bit 16x32 A/B WMMA fragment
// (CDNA5 ISA: lanes 0-15 -> K {0..7,16..23}, lanes 16-31 -> K {8..15,24..31})
__device__ __forceinline__ int kpat(int lane, int e) {
    int base = (lane & 16) ? 8 : 0;
    return base + e + ((e >= 8) ? 8 : 0);
}

__device__ __forceinline__ u16b f2bf(float x) {
    unsigned int u = __float_as_uint(x);
    unsigned int r = u + 0x7FFFu + ((u >> 16) & 1u);
    return (u16b)(r >> 16);
}

__device__ __forceinline__ float mishf(float x) {
    float sp = (x > 20.f) ? x : log1pf(__expf(x));
    return x * tanhf(sp);
}

// ---------------------------------------------------------------------------
// Pre-swizzle a row-major f32 (R x K) matrix into bf16 WMMA fragment order.
// dst layout: [(tile*Ksteps + ks)*512 + lane*16 + e]  -> 32B contiguous/lane.
// ---------------------------------------------------------------------------
__global__ void swizzle_frag(const float* __restrict__ src, u16b* __restrict__ dst,
                             int Ksteps, int Ktot, int total) {
    int gid = blockIdx.x * blockDim.x + threadIdx.x;
    if (gid >= total) return;
    int e    = gid & 15;
    int lane = (gid >> 4) & 31;
    int t    = gid >> 9;          // tile*Ksteps + ks
    int ks   = t % Ksteps;
    int tile = t / Ksteps;
    int row  = tile * 16 + (lane & 15);
    int K    = ks * 32 + kpat(lane, e);
    dst[gid] = f2bf(src[row * Ktot + K]);
}

// ---------------------------------------------------------------------------
// Kernel 1: x = mish(BN(trees @ W1^T + b1)) -> bf16 (B,4096) == (B,32,128)
// one 16x16 tile per wave, K = 512 in 16 WMMA steps.
// ---------------------------------------------------------------------------
__global__ void __launch_bounds__(256) gemm_lin_bn_mish(
    const u16b* __restrict__ treesfrag, const u16b* __restrict__ w1frag,
    const float* __restrict__ b1, const float* __restrict__ gamma,
    const float* __restrict__ beta, const float* __restrict__ rmean,
    const float* __restrict__ rvar, u16b* __restrict__ x0)
{
    const int lane  = threadIdx.x & 31;
    const int wave  = threadIdx.x >> 5;
    const int ntile = blockIdx.x * 8 + wave;   // 0..255 over 4096 features
    const int mtile = blockIdx.y;              // 0..127 over 2048 rows
    const int KS = 16;

    v8f acc = {};
#pragma unroll
    for (int ks = 0; ks < KS; ++ks) {
        Frag a, b;
        const v4u* ap = (const v4u*)(treesfrag + (mtile * KS + ks) * 512 + lane * 16);
        a.q[0] = ap[0]; a.q[1] = ap[1];
        const v4u* bp = (const v4u*)(w1frag + (ntile * KS + ks) * 512 + lane * 16);
        b.q[0] = bp[0]; b.q[1] = bp[1];
        acc = __builtin_amdgcn_wmma_f32_16x16x32_bf16(
            false, a.v, false, b.v, (short)0, acc, false, false);
    }

    const int j  = ntile * 16 + (lane & 15);
    const float sc = gamma[j] * rsqrtf(rvar[j] + 1e-5f);
    const float sh = beta[j] - rmean[j] * sc;
    const float bb = b1[j];
#pragma unroll
    for (int r = 0; r < 8; ++r) {
        int row = mtile * 16 + r + ((lane & 16) ? 8 : 0);
        float x = acc[r] + bb;
        x = x * sc + sh;
        x0[row * 4096 + j] = f2bf(mishf(x));
    }
}

// ---------------------------------------------------------------------------
// Fused per-sample BinaryTreeConv + TreeLayerNorm + Mish layer.
// GEMM: out[cout x 127] = W[cout x 3cin] @ G[3cin x 127],
//       G[3c+k][m] = in[c][ idx[3m+k] ]   (tree gather from LDS)
// 32 waves: ntile = wave&7 (8 column tiles), group = wave>>3 (4 row groups).
// The ks loop is FULLY UNROLLED so K, c=K/3, k=K%3 and the n0/n1/n2 select
// are compile-time: each gathered element is one ds_load_u16 at
// (const base + node*2) -- no runtime integer division in the hot loop.
// ---------------------------------------------------------------------------
template<int CIN, int COUT, bool LAST>
__device__ __forceinline__ void tree_layer(
    const u16b* __restrict__ wfrag, const float* __restrict__ cb,
    const int* lds_idx, const u16b* in_lds, u16b* out_lds,
    float* red, float* gout, int tid)
{
    constexpr int KSTEPS = (3 * CIN) / 32;
    constexpr int OPW    = COUT / 64;    // output row-tiles per wave
    const int lane = tid & 31;
    const int wave = tid >> 5;
    const int nt   = wave & 7;
    const int grp  = wave >> 3;

    if (tid < 2) red[tid] = 0.f;
    __syncthreads();

    const int  m     = nt * 16 + (lane & 15);
    const bool valid = (m < 127);
    int n0 = 0, n1 = 0, n2 = 0;
    if (valid) { n0 = lds_idx[3 * m]; n1 = lds_idx[3 * m + 1]; n2 = lds_idx[3 * m + 2]; }
    const int kbase = (lane & 16) ? 8 : 0;   // lane-half K offset

    v8f acc[OPW] = {};

#pragma unroll
    for (int ks = 0; ks < KSTEPS; ++ks) {
        Frag bf;
        if (valid) {
#pragma unroll
            for (int e = 0; e < 16; ++e) {
                // compile-time except kbase (uniform per lane-half) and node regs
                int K = ks * 32 + kbase + e + ((e >= 8) ? 8 : 0);
                int c = K / 3;
                int k = K - 3 * c;
                int node = (k == 0) ? n0 : ((k == 1) ? n1 : n2);
                bf.u[e] = in_lds[c * 128 + node];
            }
        } else {
#pragma unroll
            for (int e = 0; e < 16; ++e) bf.u[e] = 0;
        }
#pragma unroll
        for (int i = 0; i < OPW; ++i) {
            const int ot = grp * OPW + i;
            Frag af;
            const v4u* ap = (const v4u*)(wfrag + (ot * KSTEPS + ks) * 512 + lane * 16);
            af.q[0] = ap[0]; af.q[1] = ap[1];
            acc[i] = __builtin_amdgcn_wmma_f32_16x16x32_bf16(
                false, af.v, false, bf.v, (short)0, acc[i], false, false);
        }
    }

    // bias + per-lane partial sums for LayerNorm stats (zero node contributes 0)
    float s = 0.f, s2 = 0.f;
#pragma unroll
    for (int i = 0; i < OPW; ++i) {
        const int obase = (grp * OPW + i) * 16 + ((lane & 16) ? 8 : 0);
#pragma unroll
        for (int r = 0; r < 8; ++r) {
            float v = acc[i][r] + cb[obase + r];
            acc[i][r] = v;
            if (valid) { s += v; s2 += v * v; }
        }
    }
    atomicAdd(&red[0], s);
    atomicAdd(&red[1], s2);
    __syncthreads();

    const float n    = (float)(COUT * 128);
    const float mean = red[0] / n;
    float var = (red[1] - red[0] * mean) / (n - 1.0f);  // unbiased (ddof=1)
    const float inv = 1.0f / (sqrtf(fmaxf(var, 0.f)) + 1e-5f);

    if (valid) {
#pragma unroll
        for (int i = 0; i < OPW; ++i) {
            const int obase = (grp * OPW + i) * 16 + ((lane & 16) ? 8 : 0);
#pragma unroll
            for (int r = 0; r < 8; ++r) {
                float y = mishf((acc[i][r] - mean) * inv);
                int o = obase + r;
                if constexpr (LAST) gout[o * 128 + m + 1] = y;
                else                out_lds[o * 128 + m + 1] = f2bf(y);
            }
        }
    }
    // prepended zero node: value mish((0-mean)*inv) at node 0 of every channel
    if (tid < COUT) {
        float v0 = mishf((0.f - mean) * inv);
        if constexpr (LAST) gout[tid * 128] = v0;
        else                out_lds[tid * 128] = f2bf(v0);
    }
    __syncthreads();
}

__global__ void __launch_bounds__(1024) tree_stack(
    const u16b* __restrict__ x0, const int* __restrict__ idxg,
    const u16b* __restrict__ wf0, const u16b* __restrict__ wf1,
    const u16b* __restrict__ wf2, const u16b* __restrict__ wf3,
    const float* __restrict__ cb0, const float* __restrict__ cb1,
    const float* __restrict__ cb2, const float* __restrict__ cb3,
    float* __restrict__ out)
{
    extern __shared__ char smem[];
    int*   lds_idx = (int*)smem;                       // 384 ints
    float* red     = (float*)(smem + 1536);            // 2 floats (+pad)
    u16b*  r0      = (u16b*)(smem + 1552);             // 64 KB region
    u16b*  r1      = (u16b*)(smem + 1552 + 65536);     // 64 KB region

    const int tid = threadIdx.x;
    const int b   = blockIdx.x;

    for (int t = tid; t < 381; t += 1024) lds_idx[t] = idxg[b * 381 + t];
    // stage this sample's (32,128) bf16 activation into LDS
    const unsigned int* src = (const unsigned int*)(x0 + (size_t)b * 4096);
    unsigned int* dst = (unsigned int*)r0;
    for (int t = tid; t < 2048; t += 1024) dst[t] = src[t];
    __syncthreads();

    tree_layer< 32,  64, false>(wf0, cb0, lds_idx, r0, r1, red, nullptr, tid);
    tree_layer< 64, 128, false>(wf1, cb1, lds_idx, r1, r0, red, nullptr, tid);
    tree_layer<128, 256, false>(wf2, cb2, lds_idx, r0, r1, red, nullptr, tid);
    tree_layer<256, 512, true >(wf3, cb3, lds_idx, r1, r0, red,
                                out + (size_t)b * 65536, tid);
}

// ---------------------------------------------------------------------------
extern "C" void kernel_launch(void* const* d_in, const int* in_sizes, int n_in,
                              void* d_out, int out_size, void* d_ws, size_t ws_size,
                              hipStream_t stream) {
    (void)in_sizes; (void)n_in; (void)out_size; (void)ws_size;
    const float* trees  = (const float*)d_in[0];
    const int*   indexes= (const int*)  d_in[1];
    const float* W1     = (const float*)d_in[2];
    const float* b1     = (const float*)d_in[3];
    const float* gamma  = (const float*)d_in[4];
    const float* beta   = (const float*)d_in[5];
    const float* rmean  = (const float*)d_in[6];
    const float* rvar   = (const float*)d_in[7];
    const float* cw[4]  = {(const float*)d_in[8],  (const float*)d_in[10],
                           (const float*)d_in[12], (const float*)d_in[14]};
    const float* cb[4]  = {(const float*)d_in[9],  (const float*)d_in[11],
                           (const float*)d_in[13], (const float*)d_in[15]};

    u16b* ws16 = (u16b*)d_ws;
    size_t off = 0;
    u16b* treesfrag = ws16 + off; off += 2048ull * 512;
    u16b* w1frag    = ws16 + off; off += 4096ull * 512;
    u16b* wfrag[4];
    const int cins[4]  = {32, 64, 128, 256};
    const int couts[4] = {64, 128, 256, 512};
    for (int i = 0; i < 4; ++i) {
        wfrag[i] = ws16 + off;
        off += (size_t)couts[i] * 3 * cins[i];
    }
    u16b* x0 = ws16 + off; off += 2048ull * 4096;

    // fragment-order pre-swizzles (bf16)
    auto launch_sw = [&](const float* src, u16b* dst, int R, int K) {
        int Ksteps = K / 32;
        int total  = (R / 16) * Ksteps * 512;
        hipLaunchKernelGGL(swizzle_frag, dim3((total + 255) / 256), dim3(256), 0, stream,
                           src, dst, Ksteps, K, total);
    };
    launch_sw(trees, treesfrag, 2048, 512);
    launch_sw(W1,    w1frag,    4096, 512);
    for (int i = 0; i < 4; ++i) launch_sw(cw[i], wfrag[i], couts[i], 3 * cins[i]);

    // Linear + BN + Mish
    hipLaunchKernelGGL(gemm_lin_bn_mish, dim3(32, 128), dim3(256), 0, stream,
                       treesfrag, w1frag, b1, gamma, beta, rmean, rvar, x0);

    // Fused 4-layer tree-conv stack, one workgroup per sample, 132.6 KB LDS
    const size_t shmem = 1552 + 2 * 65536;
    hipFuncSetAttribute(reinterpret_cast<const void*>(tree_stack),
                        hipFuncAttributeMaxDynamicSharedMemorySize, (int)shmem);
    hipLaunchKernelGGL(tree_stack, dim3(2048), dim3(1024), shmem, stream,
                       x0, indexes, wfrag[0], wfrag[1], wfrag[2], wfrag[3],
                       cb[0], cb[1], cb[2], cb[3], (float*)d_out);
}
